// GATLayer_32195074851466
// MI455X (gfx1250) — matrix operand order repaired and checked by reading, compiled, not run on hardware
//
#include <hip/hip_runtime.h>
#include <hip/hip_bf16.h>
#include <stdint.h>

#define D 128
#define LDS_PAIR_STRIDE 68   // uints (bf16 pairs) per LDS row: 64 data + 4 pad (kills bank conflicts)

typedef __attribute__((ext_vector_type(16))) __bf16        v16bf;
typedef __attribute__((ext_vector_type(8)))  float         v8f;
typedef __attribute__((ext_vector_type(8)))  unsigned int  u32x8;
typedef __attribute__((ext_vector_type(4)))  unsigned int  u32x4;
typedef __attribute__((ext_vector_type(4)))  float         f32x4;

// gcc-style vector to match the async-to-LDS builtin's parameter type
typedef int i32x4v __attribute__((vector_size(16)));
typedef __attribute__((address_space(1))) i32x4v GInt4;    // global int4
typedef __attribute__((address_space(3))) i32x4v LInt4;    // LDS int4

#if defined(__has_builtin)
#if __has_builtin(__builtin_amdgcn_global_load_async_to_lds_b128) && \
    __has_builtin(__builtin_amdgcn_s_wait_asynccnt)
#define HAVE_ASYNC_LDS 1
#endif
#endif

__device__ __forceinline__ unsigned int pack_bf16(float lo, float hi) {
  unsigned int a = __float_as_uint(lo) + 0x8000u;   // ~RNE truncation to bf16
  unsigned int b = __float_as_uint(hi) + 0x8000u;
  return (b & 0xFFFF0000u) | (a >> 16);
}

// ---------------------------------------------------------------------------
// Kernel 1: h = x @ W^T via bf16 WMMA, fused s_src = h@a[:D], s_tgt = h@a[D:]
// One wave per 16-row strip of x; 8 waves (256 threads) per block.
// W staged to LDS (f32 chunks via async-to-LDS when available), packed to bf16.
// ---------------------------------------------------------------------------
__launch_bounds__(256)
__global__ void gat_gemm_kernel(const float* __restrict__ x,
                                const float* __restrict__ W,
                                const float* __restrict__ a,
                                float* __restrict__ h,
                                float* __restrict__ s_src,
                                float* __restrict__ s_tgt,
                                int nStrips)
{
  __shared__ float        wstage[32 * D];               // 16 KB staging (32 rows of W)
  __shared__ unsigned int wpack[D * LDS_PAIR_STRIDE];   // 34 KB packed bf16 W

  const int tid = threadIdx.x;

  // ---- stage W into LDS as packed bf16, 32 rows at a time ----
  for (int chunk = 0; chunk < 4; ++chunk) {
    const float* wsrc = W + (size_t)chunk * 32 * D;
#if defined(HAVE_ASYNC_LDS)
    #pragma unroll
    for (int i = 0; i < 4; ++i) {
      int off = (tid + i * 256) * 4;                    // float index within 4096-float chunk
      __builtin_amdgcn_global_load_async_to_lds_b128(
          (GInt4*)(uintptr_t)(wsrc + off),
          (LInt4*)(unsigned int)(uintptr_t)(wstage + off),
          0, 0);
    }
    __builtin_amdgcn_s_wait_asynccnt(0);
#else
    #pragma unroll
    for (int i = 0; i < 4; ++i) {
      int off = (tid + i * 256) * 4;
      *(f32x4*)(wstage + off) = *(const f32x4*)(wsrc + off);
    }
#endif
    __syncthreads();
    #pragma unroll
    for (int i = 0; i < 8; ++i) {
      int p   = tid + i * 256;                          // bf16-pair index within chunk
      int row = p >> 6;                                 // 64 pairs per row
      int cp  = p & 63;
      wpack[(chunk * 32 + row) * LDS_PAIR_STRIDE + cp] =
          pack_bf16(wstage[row * D + 2 * cp], wstage[row * D + 2 * cp + 1]);
    }
    __syncthreads();
  }

  const int lane  = tid & 31;
  const int wv    = tid >> 5;
  const int strip = blockIdx.x * 8 + wv;
  if (strip >= nStrips) return;                         // uniform per wave; EXEC stays full

  const int rit = lane & 15;                            // row (A) / col (B,C) within tile
  const int hi  = lane >> 4;                            // K-half selector

  // ---- A fragments: 16 rows x 128 K, bf16-packed per the 16-bit A layout ----
  u32x8 afrag[4];
  {
    const float* xrow = x + (size_t)(strip * 16 + rit) * D;
    #pragma unroll
    for (int kc = 0; kc < 4; ++kc) {
      int k0 = kc * 32 + hi * 8;                        // K {0..7} / {8..15}
      int k1 = k0 + 16;                                 // K {16..23} / {24..31}
      f32x4 f0 = *(const f32x4*)(xrow + k0);
      f32x4 f1 = *(const f32x4*)(xrow + k0 + 4);
      f32x4 f2 = *(const f32x4*)(xrow + k1);
      f32x4 f3 = *(const f32x4*)(xrow + k1 + 4);
      u32x8 u;
      u[0] = pack_bf16(f0[0], f0[1]); u[1] = pack_bf16(f0[2], f0[3]);
      u[2] = pack_bf16(f1[0], f1[1]); u[3] = pack_bf16(f1[2], f1[3]);
      u[4] = pack_bf16(f2[0], f2[1]); u[5] = pack_bf16(f2[2], f2[3]);
      u[6] = pack_bf16(f3[0], f3[1]); u[7] = pack_bf16(f3[2], f3[3]);
      afrag[kc] = u;
    }
  }

  float ssrc[8] = {0.f, 0.f, 0.f, 0.f, 0.f, 0.f, 0.f, 0.f};
  float stgt[8] = {0.f, 0.f, 0.f, 0.f, 0.f, 0.f, 0.f, 0.f};

  for (int t = 0; t < 8; ++t) {                         // 8 column tiles of 16
    v8f c = {};
    const unsigned int* bbase =
        wpack + (t * 16 + rit) * LDS_PAIR_STRIDE + hi * 4;
    #pragma unroll
    for (int kc = 0; kc < 4; ++kc) {
      const u32x4* bp = (const u32x4*)(bbase + kc * 16);
      u32x4 b0 = bp[0];
      u32x4 b1 = bp[2];
      u32x8 bu;
      bu[0] = b0[0]; bu[1] = b0[1]; bu[2] = b0[2]; bu[3] = b0[3];
      bu[4] = b1[0]; bu[5] = b1[1]; bu[6] = b1[2]; bu[7] = b1[3];
      c = __builtin_amdgcn_wmma_f32_16x16x32_bf16(
              false, __builtin_bit_cast(v16bf, afrag[kc]),
              false, __builtin_bit_cast(v16bf, bu),
              (short)0, c, false, false);
    }

    // epilogue: store h tile, accumulate attention-logit partials
    int   col = t * 16 + rit;
    float av  = a[col];
    float bv  = a[D + col];
    float* hp = h + ((size_t)(strip * 16 + hi * 8)) * D + col;
    #pragma unroll
    for (int r = 0; r < 8; ++r) {
      float cv = c[r];
      hp[(size_t)r * D] = cv;
      ssrc[r] = __builtin_fmaf(cv, av, ssrc[r]);
      stgt[r] = __builtin_fmaf(cv, bv, stgt[r]);
    }
  }

  // reduce per-row partials across the 16 lanes that share each C row
  #pragma unroll
  for (int m = 1; m < 16; m <<= 1) {
    #pragma unroll
    for (int r = 0; r < 8; ++r) {
      ssrc[r] += __shfl_xor(ssrc[r], m, 32);
      stgt[r] += __shfl_xor(stgt[r], m, 32);
    }
  }
  if (rit == 0) {
    int base = strip * 16 + hi * 8;
    #pragma unroll
    for (int r = 0; r < 8; ++r) {
      s_src[base + r] = ssrc[r];
      s_tgt[base + r] = stgt[r];
    }
  }
}

// ---------------------------------------------------------------------------
// Kernel 2: per-edge exp(leaky_relu) + atomic segment sum over tgt
// ---------------------------------------------------------------------------
__global__ void gat_edge_sum_kernel(const int* __restrict__ ei,
                                    const float* __restrict__ s_src,
                                    const float* __restrict__ s_tgt,
                                    float* __restrict__ att_sum,
                                    int E)
{
  int e = blockIdx.x * blockDim.x + threadIdx.x;
  if (e >= E) return;
  int src = ei[e];
  int tgt = ei[E + e];
  float v  = s_src[src] + s_tgt[tgt];
  float lr = v > 0.f ? v : 0.2f * v;
  unsafeAtomicAdd(&att_sum[tgt], __expf(lr));
}

// ---------------------------------------------------------------------------
// Kernel 3: one wave32 per edge; gather h[src] (float4/lane), scatter-add
// att_norm * h[src] into h'[tgt] with f32 atomics.
// ---------------------------------------------------------------------------
__global__ void gat_edge_scatter_kernel(const int* __restrict__ ei,
                                        const float* __restrict__ h,
                                        const float* __restrict__ s_src,
                                        const float* __restrict__ s_tgt,
                                        const float* __restrict__ att_sum,
                                        float* __restrict__ hprime,
                                        int E)
{
  long long gid = (long long)blockIdx.x * blockDim.x + threadIdx.x;
  int e    = (int)(gid >> 5);
  int lane = (int)(gid & 31);
  if (e >= E) return;
  int src = ei[e];
  int tgt = ei[E + e];
  float v   = s_src[src] + s_tgt[tgt];
  float lr  = v > 0.f ? v : 0.2f * v;
  float w   = __expf(lr) / (att_sum[tgt] + 1e-8f);
  f32x4 hv  = *(const f32x4*)(h + (size_t)src * D + lane * 4);
  float* dst = hprime + (size_t)tgt * D + lane * 4;
  unsafeAtomicAdd(dst + 0, hv[0] * w);
  unsafeAtomicAdd(dst + 1, hv[1] * w);
  unsafeAtomicAdd(dst + 2, hv[2] * w);
  unsafeAtomicAdd(dst + 3, hv[3] * w);
}

// ---------------------------------------------------------------------------
// Kernel 4: out = elu(h' + x), in place on d_out
// ---------------------------------------------------------------------------
__global__ void gat_finalize_kernel(const float* __restrict__ x,
                                    float* __restrict__ out,
                                    int n)
{
  int i = blockIdx.x * blockDim.x + threadIdx.x;
  if (i >= n) return;
  float v = out[i] + x[i];
  out[i] = v > 0.f ? v : (__expf(v) - 1.f);
}

__global__ void zero_kernel(float* __restrict__ p, size_t n)
{
  size_t i      = (size_t)blockIdx.x * blockDim.x + threadIdx.x;
  size_t stride = (size_t)gridDim.x * blockDim.x;
  for (; i < n; i += stride) p[i] = 0.f;
}

extern "C" void kernel_launch(void* const* d_in, const int* in_sizes, int n_in,
                              void* d_out, int out_size, void* d_ws, size_t ws_size,
                              hipStream_t stream)
{
  const float* x  = (const float*)d_in[0];   // [N, D]
  const float* W  = (const float*)d_in[1];   // [D, D]
  const float* a  = (const float*)d_in[2];   // [2*D]
  const int*   ei = (const int*)d_in[3];     // [2, E]
  float* out = (float*)d_out;                // [N, D]

  const int N = in_sizes[0] / D;             // 100000
  const int E = in_sizes[3] / 2;             // 1600000

  // workspace layout
  float* h      = (float*)d_ws;              // N*D
  float* s_src  = h + (size_t)N * D;         // N
  float* s_tgt  = s_src + N;                 // N
  float* attsum = s_tgt + N;                 // N

  zero_kernel<<<2048, 256, 0, stream>>>(out, (size_t)N * D);
  zero_kernel<<<256, 256, 0, stream>>>(attsum, (size_t)N);

  int nStrips    = (N + 15) / 16;            // 6250
  int gemmBlocks = (nStrips + 7) / 8;
  gat_gemm_kernel<<<gemmBlocks, 256, 0, stream>>>(x, W, a, h, s_src, s_tgt, nStrips);

  gat_edge_sum_kernel<<<(E + 255) / 256, 256, 0, stream>>>(ei, s_src, s_tgt, attsum, E);

  long long tot = (long long)E * 32;
  gat_edge_scatter_kernel<<<(unsigned)((tot + 255) / 256), 256, 0, stream>>>(
      ei, h, s_src, s_tgt, attsum, out, E);

  gat_finalize_kernel<<<(N * D + 255) / 256, 256, 0, stream>>>(x, out, N * D);
}
